// SemanticRematcher_21517786153375
// MI455X (gfx1250) — matrix operand order repaired and checked by reading, not compiled, not run on hardware
//
#include <hip/hip_runtime.h>
#include <hip/hip_bf16.h>
#include <stdint.h>

// ---- CDNA5 WMMA types (wave32) ----
typedef __attribute__((ext_vector_type(16))) _Float16 v16h;
typedef __attribute__((ext_vector_type(16))) __bf16   v16bf;
typedef __attribute__((ext_vector_type(8)))  float    v8f;

#define BN 1024          // batch
#define DN 1024          // feature dim
#define REG_INV 10.0f    // 1/REG
#define EPS_F 1e-8f
#define THRESH 1e-3f
#define MAX_ITER 100

union BFrag { uint4 q[2]; v16bf v; };
union HFrag { uint4 q[2]; v16h  v; };

__device__ inline unsigned short f2bf(float f) {
    unsigned int u = __float_as_uint(f);
    unsigned int r = u + 0x7FFFu + ((u >> 16) & 1u);   // round-to-nearest-even
    return (unsigned short)(r >> 16);
}

// ------------------------------------------------------------------
// Kernel 1: L2-normalize rows of visual/textual feats -> bf16 rows.
// One wave per row, 2048 rows total.
// ------------------------------------------------------------------
__global__ __launch_bounds__(256) void k_normalize(
        const float* __restrict__ vis, const float* __restrict__ txt,
        unsigned short* __restrict__ vnb, unsigned short* __restrict__ tnb) {
    int lane = threadIdx.x & 31;
    int wave = threadIdx.x >> 5;
    int row  = blockIdx.x * 8 + wave;                       // 0..2047
    const float* src = (row < BN) ? (vis + (size_t)row * DN)
                                  : (txt + (size_t)(row - BN) * DN);
    unsigned short* dst = (row < BN) ? (vnb + (size_t)row * DN)
                                     : (tnb + (size_t)(row - BN) * DN);
    float x[32];
    float ss = 0.f;
#pragma unroll
    for (int i = 0; i < 32; i++) { x[i] = src[i * 32 + lane]; ss = fmaf(x[i], x[i], ss); }
#pragma unroll
    for (int m = 16; m >= 1; m >>= 1) ss += __shfl_xor(ss, m, 32);
    float inv = rsqrtf(ss);
#pragma unroll
    for (int i = 0; i < 32; i++) dst[i * 32 + lane] = f2bf(x[i] * inv);
}

// ------------------------------------------------------------------
// Kernel 2: sim = vn @ tn^T via v_wmma_f32_16x16x32_bf16.
// One wave per 16x16 output tile; K-loop of 32 steps.
// ------------------------------------------------------------------
__global__ __launch_bounds__(256) void k_sim(
        const unsigned short* __restrict__ vnb,
        const unsigned short* __restrict__ tnb,
        float* __restrict__ sim) {
    int lane = threadIdx.x & 31;
    int wave = threadIdx.x >> 5;
    int w  = blockIdx.x * 8 + wave;                         // 0..4095 tiles
    int tm = w >> 6, tn = w & 63;
    int lo = lane & 15, hi = lane >> 4;

    const unsigned short* arow = vnb + (size_t)(tm * 16 + lo) * DN;  // A: M=lo
    const unsigned short* brow = tnb + (size_t)(tn * 16 + lo) * DN;  // B: N=lo (tn^T)

    v8f acc = {0.f, 0.f, 0.f, 0.f, 0.f, 0.f, 0.f, 0.f};
#pragma unroll 4
    for (int kk = 0; kk < DN; kk += 32) {
        BFrag a, b;
        // A 16x32 bf16: elems 0-15 -> K=kk+hi*16..+15 (contiguous 32B)
        const uint4* ap = reinterpret_cast<const uint4*>(arow + kk + hi * 16);
        a.q[0] = ap[0];
        a.q[1] = ap[1];
        // B 32x16 bf16: elems 0-15 -> K=kk+hi*16..+15 (contiguous 32B)
        const uint4* bp = reinterpret_cast<const uint4*>(brow + kk + hi * 16);
        b.q[0] = bp[0];
        b.q[1] = bp[1];
        if (kk + 32 < DN) {
            __builtin_prefetch(arow + kk + 32 + hi * 8, 0, 1);
            __builtin_prefetch(brow + kk + 32 + hi * 16, 0, 1);
        }
        acc = __builtin_amdgcn_wmma_f32_16x16x32_bf16(false, a.v, false, b.v,
                                                      (short)0, acc, false, false);
    }
    float* out = sim + (size_t)(tm * 16 + hi * 8) * BN + tn * 16 + lo;
#pragma unroll
    for (int r = 0; r < 8; r++) out[(size_t)r * BN] = acc[r];
}

// ------------------------------------------------------------------
// Kernel 3: cost MLP on 1M sim values + K = exp(-cost/REG).
// Per wave: 16 rows/group; layer2 (128->64) and layer3 (64->32) via
// v_wmma_f32_16x16x32_f16 with weight B-fragments resident in VGPRs.
// Layer2 runs ks-outer so a single A fragment is live at a time and
// 4 consecutive WMMAs share the same A operand (HW reuse-friendly).
// Layer2->layer3 transpose bounced through LDS (ds b128 A-loads).
// ------------------------------------------------------------------
__global__ __launch_bounds__(256) void k_cost(
        const float* __restrict__ sim,
        const float* __restrict__ W1, const float* __restrict__ b1,
        const float* __restrict__ W2, const float* __restrict__ b2,
        const float* __restrict__ W3, const float* __restrict__ b3,
        const float* __restrict__ W4, const float* __restrict__ b4,
        float* __restrict__ cost, float* __restrict__ Kmat) {
    __shared__ float w1s[128], b1s[128], b2s[64], b3s[32], w4s[32];
    __shared__ float b4s;
    __shared__ _Float16 __align__(16) h2s[8][16 * 64];     // per-wave 16x64 scratch

    int t = threadIdx.x;
    if (t < 128) w1s[t] = W1[t]; else b1s[t - 128] = b1[t - 128];
    if (t < 64) b2s[t] = b2[t];
    else if (t < 96) b3s[t - 64] = b3[t - 64];
    else if (t < 128) w4s[t - 96] = W4[t - 96];
    if (t == 0) b4s = b4[0];
    __syncthreads();

    int lane = t & 31, wv = t >> 5;
    int lo = lane & 15, hi = lane >> 4;
    int waveG = blockIdx.x * 8 + wv;                        // 0..2047

    // Persistent B fragments: W2 [128x64] -> 4 k-steps x 4 n-tiles
    v16h w2f[4][4];
#pragma unroll
    for (int ks = 0; ks < 4; ks++)
#pragma unroll
        for (int nt = 0; nt < 4; nt++) {
            v16h f;
#pragma unroll
            for (int e = 0; e < 16; e++) {
                int k = ks * 32 + hi * 16 + e;              // k order used here for B and A: K=(l/16)*16+idx
                f[e] = (_Float16)W2[k * 64 + nt * 16 + lo];
            }
            w2f[ks][nt] = f;
        }
    // W3 [64x32] -> 2 k-steps x 2 n-tiles
    v16h w3f[2][2];
#pragma unroll
    for (int ks = 0; ks < 2; ks++)
#pragma unroll
        for (int nt = 0; nt < 2; nt++) {
            v16h f;
#pragma unroll
            for (int e = 0; e < 16; e++) {
                int k = ks * 32 + hi * 16 + e;
                f[e] = (_Float16)W3[k * 32 + nt * 16 + lo];
            }
            w3f[ks][nt] = f;
        }

    for (int it = 0; it < 32; it++) {
        int group = it * 2048 + waveG;                      // 0..65535
        int base  = group * 16;
        float x = sim[base + lo];                           // row M=lo scalar input

        // Layer 2 accumulators, pre-loaded with bias
        v8f acc2[4];
#pragma unroll
        for (int nt = 0; nt < 4; nt++) {
            float bb = b2s[nt * 16 + lo];
#pragma unroll
            for (int r = 0; r < 8; r++) acc2[nt][r] = bb;
        }

        // Layer 1 fused into layer-2 k-loop: build one A fragment per
        // k-step, feed 4 WMMAs sharing that A operand.
#pragma unroll
        for (int ks = 0; ks < 4; ks++) {
            v16h a1;
#pragma unroll
            for (int e = 0; e < 16; e++) {
                int k = ks * 32 + hi * 16 + e;
                float h = fmaf(x, w1s[k], b1s[k]);
                a1[e] = (_Float16)fmaxf(h, 0.f);
            }
#pragma unroll
            for (int nt = 0; nt < 4; nt++)
                acc2[nt] = __builtin_amdgcn_wmma_f32_16x16x32_f16(false, a1, false,
                        w2f[ks][nt], (short)0, acc2[nt], false, false);
        }

        // relu(h2) -> LDS (C-layout store, row-major 16x64 f16)
#pragma unroll
        for (int nt = 0; nt < 4; nt++)
#pragma unroll
            for (int r = 0; r < 8; r++)
                h2s[wv][(r + hi * 8) * 64 + nt * 16 + lo] =
                    (_Float16)fmaxf(acc2[nt][r], 0.f);

        // Reload h2 as layer-3 A fragments (transpose via LDS, ds b128)
        v16h a3[2];
#pragma unroll
        for (int ks = 0; ks < 2; ks++) {
            HFrag f;
            const uint4* p = reinterpret_cast<const uint4*>(
                &h2s[wv][lo * 64 + ks * 32 + hi * 16]);
            f.q[0] = p[0];
            f.q[1] = p[1];
            a3[ks] = f.v;
        }

        // Layer 3: h3 = relu(h2 @ W3 + b3), 4 WMMAs
        v8f h3[2];
#pragma unroll
        for (int nt = 0; nt < 2; nt++) {
            float bb = b3s[nt * 16 + lo];
            v8f acc;
#pragma unroll
            for (int r = 0; r < 8; r++) acc[r] = bb;
#pragma unroll
            for (int ks = 0; ks < 2; ks++)
                acc = __builtin_amdgcn_wmma_f32_16x16x32_f16(false, a3[ks], false,
                        w3f[ks][nt], (short)0, acc, false, false);
#pragma unroll
            for (int r = 0; r < 8; r++) acc[r] = fmaxf(acc[r], 0.f);
            h3[nt] = acc;
        }

        // Layer 4: dot over N=32 (cross-lane reduce in 16-lane halves)
        float s[8];
        float wA = w4s[lo], wB = w4s[16 + lo];
#pragma unroll
        for (int r = 0; r < 8; r++) s[r] = h3[0][r] * wA + h3[1][r] * wB;
#pragma unroll
        for (int m = 1; m < 16; m <<= 1)
#pragma unroll
            for (int r = 0; r < 8; r++) s[r] += __shfl_xor(s[r], m, 32);

        if (lo < 8) {
            float val = s[0];
#pragma unroll
            for (int r = 1; r < 8; r++) val = (lo == r) ? s[r] : val;
            float z   = val + b4s;
            float sg  = 1.f / (1.f + __expf(-z));
            float cst = 1.f - sg;
            int idx = base + lo + hi * 8;                   // m = r + 8*(l/16)
            cost[idx] = cst;
            Kmat[idx] = __expf(-cst * REG_INV);
        }
    }
}

// ------------------------------------------------------------------
// Sinkhorn (replicates scan + done-freeze semantics, deterministic)
// ------------------------------------------------------------------
__global__ void k_sink_init(float* u, float* v, float* du2, int* done) {
    int i = blockIdx.x * blockDim.x + threadIdx.x;
    if (i < BN) { u[i] = 1.f / BN; v[i] = 1.f / BN; du2[i] = 0.f; }
    if (i == 0) *done = 0;
}

__global__ void k_sink_v(const float* __restrict__ K, const float* __restrict__ u,
                         float* __restrict__ v, const int* __restrict__ done) {
    if (*done) return;
    int j = blockIdx.x * blockDim.x + threadIdx.x;          // 1024 threads, coalesced
    float sum = 0.f;
    for (int i = 0; i < BN; i++) sum = fmaf(K[(size_t)i * BN + j], u[i], sum);
    v[j] = (1.f / BN) / (sum + EPS_F);
}

__global__ __launch_bounds__(256) void k_sink_u(
        const float* __restrict__ K, const float* __restrict__ v,
        float* __restrict__ u, float* __restrict__ du2,
        const int* __restrict__ done) {
    if (*done) return;
    int lane = threadIdx.x & 31;
    int i = blockIdx.x * 8 + (threadIdx.x >> 5);            // one wave per row
    const float* row = K + (size_t)i * BN;
    float sum = 0.f;
    for (int j = lane; j < BN; j += 32) sum = fmaf(row[j], v[j], sum);
#pragma unroll
    for (int m = 16; m >= 1; m >>= 1) sum += __shfl_xor(sum, m, 32);
    if (lane == 0) {
        float un = (1.f / BN) / (sum + EPS_F);
        float d  = un - u[i];
        du2[i] = d * d;
        u[i]   = un;
    }
}

__global__ void k_sink_fin(const float* __restrict__ du2, int* done) {
    __shared__ float red[256];
    float s = 0.f;
    for (int i = threadIdx.x; i < BN; i += 256) s += du2[i];
    red[threadIdx.x] = s;
    __syncthreads();
    for (int st = 128; st > 0; st >>= 1) {
        if (threadIdx.x < st) red[threadIdx.x] += red[threadIdx.x + st];
        __syncthreads();
    }
    if (threadIdx.x == 0) { if (!*done && sqrtf(red[0]) < THRESH) *done = 1; }
}

__global__ __launch_bounds__(256) void k_transport(
        const float* __restrict__ K, const float* __restrict__ u,
        const float* __restrict__ v, float* __restrict__ T) {
    int idx = blockIdx.x * blockDim.x + threadIdx.x;
    int i = idx >> 10, j = idx & 1023;
    T[idx] = u[i] * K[idx] * v[j];
}

// ------------------------------------------------------------------
extern "C" void kernel_launch(void* const* d_in, const int* in_sizes, int n_in,
                              void* d_out, int out_size, void* d_ws, size_t ws_size,
                              hipStream_t stream) {
    (void)in_sizes; (void)n_in; (void)out_size; (void)ws_size;
    const float* vis = (const float*)d_in[0];
    const float* txt = (const float*)d_in[1];
    const float* W1  = (const float*)d_in[2];
    const float* b1  = (const float*)d_in[3];
    const float* W2  = (const float*)d_in[4];
    const float* b2  = (const float*)d_in[5];
    const float* W3  = (const float*)d_in[6];
    const float* b3  = (const float*)d_in[7];
    const float* W4  = (const float*)d_in[8];
    const float* b4  = (const float*)d_in[9];
    // d_in[10] = uncertain_mask (all-True) -> full-batch Sinkhorn, ignored.

    float* T    = (float*)d_out;                            // transport [1M]
    float* cost = (float*)d_out + 1024 * 1024;              // cost      [1M]
    float* sim  = (float*)d_out + 2 * 1024 * 1024;          // sim       [1M]

    char* ws = (char*)d_ws;
    unsigned short* vnb = (unsigned short*)ws;                               // 2 MB
    unsigned short* tnb = (unsigned short*)(ws + (size_t)2 * 1024 * 1024);   // 2 MB
    float* Km  = (float*)(ws + (size_t)4 * 1024 * 1024);                     // 4 MB
    float* u   = (float*)(ws + (size_t)8 * 1024 * 1024);
    float* v   = (float*)(ws + (size_t)8 * 1024 * 1024 + 4096);
    float* du2 = (float*)(ws + (size_t)8 * 1024 * 1024 + 8192);
    int*  done = (int*) (ws + (size_t)8 * 1024 * 1024 + 12288);

    k_normalize<<<256, 256, 0, stream>>>(vis, txt, vnb, tnb);
    k_sim<<<512, 256, 0, stream>>>(vnb, tnb, sim);
    k_cost<<<256, 256, 0, stream>>>(sim, W1, b1, W2, b2, W3, b3, W4, b4, cost, Km);
    k_sink_init<<<4, 256, 0, stream>>>(u, v, du2, done);
    for (int it = 0; it < MAX_ITER; it++) {
        k_sink_v<<<4, 256, 0, stream>>>(Km, u, v, done);
        k_sink_u<<<128, 256, 0, stream>>>(Km, v, u, du2, done);
        k_sink_fin<<<1, 256, 0, stream>>>(du2, done);
    }
    k_transport<<<4096, 256, 0, stream>>>(Km, u, v, T);
}
